// AttLoc_65549790871844
// MI455X (gfx1250) — matrix operand order repaired and checked by reading, not compiled, not run on hardware
//
#include <hip/hip_runtime.h>

// ---------------- problem constants ----------------
#define B_      32
#define T_      1600
#define EPROJS  512
#define DUNITS  1024
#define ATT     512
#define CH      10
#define KS      201
#define KEXT    544          // 512 enc-K + 10 conv-K + 22 zero pad (17 chunks of 32)
#define ASTRIDE 552          // padded LDS row stride (ushorts) for A
#define BSTRIDE 40           // padded LDS row stride (ushorts) for B

typedef __attribute__((ext_vector_type(16))) __bf16        bf16x16;
typedef __attribute__((ext_vector_type(8)))  float         f32x8;
typedef __attribute__((ext_vector_type(4)))  float         f32x4;
typedef __attribute__((ext_vector_type(4)))  unsigned int  u32x4;
typedef __attribute__((ext_vector_type(2)))  unsigned int  u32x2;

struct U32x8 { u32x4 a, b; };

static __device__ __forceinline__ unsigned short f2bf(float f) {
  unsigned int u = __float_as_uint(f);
  u += 0x7FFFu + ((u >> 16) & 1u);          // round-to-nearest-even
  return (unsigned short)(u >> 16);
}

// two 16B pieces at p and p+delta (ushort units) -> one 16-elem bf16 fragment
static __device__ __forceinline__ bf16x16 load_frag2(const unsigned short* p, int delta) {
  U32x8 t;
  t.a = *(const u32x4*)(p);
  t.b = *(const u32x4*)(p + delta);
  return __builtin_bit_cast(bf16x16, t);
}

// ---------------- prep: W_ext = [W_enc | W_att | 0] in bf16, [512][544] ----------------
__global__ __launch_bounds__(256)
void prep_wext_kernel(const float* __restrict__ Wenc, const float* __restrict__ Watt,
                      unsigned short* __restrict__ wext) {
  const int a = blockIdx.x;
  for (int k = threadIdx.x; k < KEXT; k += 256) {
    float v;
    if (k < 512)      v = Wenc[(size_t)a * 512 + k];
    else if (k < 522) v = Watt[(size_t)a * CH + (k - 512)];
    else              v = 0.f;
    wext[(size_t)a * KEXT + k] = f2bf(v);
  }
}

// ---------------- prep: dvec[b][a] = dec_z[b]·W_dec[a] + b_enc[a] ----------------
__global__ __launch_bounds__(256)
void prep_dvec_kernel(const float* __restrict__ dec_z, const float* __restrict__ Wdec,
                      const float* __restrict__ b_enc, float* __restrict__ dvec) {
  const int a = blockIdx.x, b = blockIdx.y, tid = threadIdx.x;
  __shared__ float red[8];
  float s = 0.f;
  for (int e = tid; e < DUNITS; e += 256)
    s += dec_z[(size_t)b * DUNITS + e] * Wdec[(size_t)a * DUNITS + e];
  for (int mask = 16; mask; mask >>= 1) s += __shfl_xor(s, mask, 32);
  if ((tid & 31) == 0) red[tid >> 5] = s;
  __syncthreads();
  if (tid == 0) {
    float t = 0.f;
    for (int i = 0; i < 8; ++i) t += red[i];
    dvec[(size_t)b * ATT + a] = t + b_enc[a];
  }
}

// ---------------- location conv: convbuf[b][t][16] (c<10 valid, rest 0) ----------------
__global__ __launch_bounds__(256)
void conv_kernel(const float* __restrict__ att_prev, const float* __restrict__ conv_w,
                 float* __restrict__ convbuf) {
  __shared__ float win[256 + 200];
  __shared__ float cw[CH * KS];
  const int b = blockIdx.y, t0 = blockIdx.x * 256, tid = threadIdx.x;
  for (int i = tid; i < 256 + 200; i += 256) {
    int pos = t0 - 100 + i;
    win[i] = (pos >= 0 && pos < T_) ? att_prev[(size_t)b * T_ + pos] : 0.f;
  }
  for (int i = tid; i < CH * KS; i += 256) cw[i] = conv_w[i];
  __syncthreads();
  const int t = t0 + tid;
  if (t < T_) {
    float s[CH];
#pragma unroll
    for (int c = 0; c < CH; ++c) s[c] = 0.f;
    for (int k = 0; k < KS; ++k) {
      float x = win[tid + k];
#pragma unroll
      for (int c = 0; c < CH; ++c) s[c] += cw[c * KS + k] * x;
    }
    float* dst = convbuf + ((size_t)b * T_ + t) * 16;
#pragma unroll
    for (int c = 0; c < CH; ++c) dst[c] = s[c];
#pragma unroll
    for (int c = CH; c < 16; ++c) dst[c] = 0.f;
  }
}

// ---------------- main fused WMMA energy kernel ----------------
// grid (25, 32), block 512 (16 waves). Wave (rt,cg): rows rt*16..+15, cols cg*128..+127.
// Double-buffered B staging: one barrier per K-chunk, next chunk's global loads
// issued before the current chunk's 8 WMMAs.
__global__ __launch_bounds__(512)
void energy_kernel(const float* __restrict__ enc,
                   const float* __restrict__ convbuf,
                   const float* __restrict__ dvec,
                   const unsigned short* __restrict__ wext,
                   const float* __restrict__ gvec,
                   float* __restrict__ e_out) {
  __shared__ __align__(16) unsigned short Abuf[64 * ASTRIDE];       // 70.7 KB
  __shared__ __align__(16) unsigned short Bbuf[2][512 * BSTRIDE];   // 80.0 KB
  __shared__ float e_part[4][64];

  const int b = blockIdx.y;
  const int t0 = blockIdx.x * 64;
  const int tid = threadIdx.x;
  const int wv = tid >> 5, lane = tid & 31;
  const int rt = wv >> 2, cg = wv & 3;
  const int lrow = lane & 15, hi = lane >> 4;

  // Stage A: 64 rows x 512 fp32 -> bf16 in LDS (cols 0..511)
  const f32x4* encb = (const f32x4*)(enc + ((size_t)b * T_ + t0) * EPROJS);
#pragma unroll
  for (int it = 0; it < 16; ++it) {
    int j = tid + it * 512;            // float4 index 0..8191
    int r = j >> 7;                    // 128 float4 per row
    int c = (j & 127) << 2;
    f32x4 v = encb[j];
    u32x2 pk;
    pk.x = (unsigned int)f2bf(v.x) | ((unsigned int)f2bf(v.y) << 16);
    pk.y = (unsigned int)f2bf(v.z) | ((unsigned int)f2bf(v.w) << 16);
    *(u32x2*)&Abuf[r * ASTRIDE + c] = pk;
  }
  // Stage A tail: conv columns 512..527, zeros 528..543
  {
    int r = tid >> 3;                  // 0..63
    int c0 = (tid & 7) << 1;           // 0,2,..,14
    const float* cb = convbuf + ((size_t)b * T_ + (t0 + r)) * 16 + c0;
    unsigned short* dst = &Abuf[r * ASTRIDE + 512 + c0];
    dst[0] = f2bf(cb[0]);
    dst[1] = f2bf(cb[1]);
    dst[16] = 0; dst[17] = 0;
  }

  // Prologue: stage B chunk 0 into buffer 0 (each thread owns column n = tid)
  const u32x4* bsrc = (const u32x4*)(wext + (size_t)tid * KEXT);
  {
    u32x4* dst = (u32x4*)&Bbuf[0][tid * BSTRIDE];
    dst[0] = bsrc[0]; dst[1] = bsrc[1]; dst[2] = bsrc[2]; dst[3] = bsrc[3];
  }
  __syncthreads();                     // covers A staging + B chunk 0

  f32x8 acc[8] = {};
  const unsigned short* arow = &Abuf[(rt * 16 + lrow) * ASTRIDE];

  for (int kc = 0; kc < 17; ++kc) {
    const int cur = kc & 1;

    // Issue next chunk's global loads early (hidden behind the 8 WMMAs below)
    u32x4 r0, r1, r2, r3;
    if (kc < 16) {
      const u32x4* s = bsrc + (kc + 1) * 2;    // (kc+1)*16 ushorts = (kc+1)*2 u32x4
      r0 = s[0]; r1 = s[1]; r2 = s[2]; r3 = s[3];
    }

    // A fragment: lanes 0-15 hold K {0..7,16..23}, lanes 16-31 hold K {8..15,24..31}
    bf16x16 af = load_frag2(arow + kc * 32 + hi * 8, 16);

#pragma unroll
    for (int tile = 0; tile < 8; ++tile) {
      int n = cg * 128 + tile * 16 + lrow;
      // B fragment: lanes 0-15 hold K {0..7,16..23}, lanes 16-31 hold K {8..15,24..31}
      bf16x16 bfg = load_frag2(&Bbuf[cur][n * BSTRIDE + hi * 8], 16);
      acc[tile] = __builtin_amdgcn_wmma_f32_16x16x32_bf16(
          false, af, false, bfg, (short)0, acc[tile], false, false);
    }

    // Store next chunk into the other buffer; single barrier per iteration
    if (kc < 16) {
      u32x4* dst = (u32x4*)&Bbuf[cur ^ 1][tid * BSTRIDE];
      dst[0] = r0; dst[1] = r1; dst[2] = r2; dst[3] = r3;
    }
    __syncthreads();
  }

  // Epilogue: +dvec, tanh, *gvec, reduce over a
  float dv[8], gv[8];
#pragma unroll
  for (int tile = 0; tile < 8; ++tile) {
    int a = cg * 128 + tile * 16 + lrow;
    dv[tile] = dvec[(size_t)b * ATT + a];
    gv[tile] = gvec[a];
  }
#pragma unroll
  for (int r = 0; r < 8; ++r) {
    float s = 0.f;
#pragma unroll
    for (int tile = 0; tile < 8; ++tile)
      s += gv[tile] * tanhf(acc[tile][r] + dv[tile]);
    s += __shfl_xor(s, 1, 32);
    s += __shfl_xor(s, 2, 32);
    s += __shfl_xor(s, 4, 32);
    s += __shfl_xor(s, 8, 32);
    if (lrow == 0) e_part[cg][rt * 16 + (hi << 3) + r] = s;  // row = r + 8*hi
  }
  __syncthreads();
  if (tid < 64)
    e_out[(size_t)b * T_ + t0 + tid] =
        e_part[0][tid] + e_part[1][tid] + e_part[2][tid] + e_part[3][tid];
}

// ---------------- softmax over T (SCALING=2 folded in) ----------------
__global__ __launch_bounds__(256)
void softmax_kernel(const float* __restrict__ e, float* __restrict__ w) {
  const int b = blockIdx.x, tid = threadIdx.x;
  __shared__ float red[8];
  __shared__ float bcast;
  float vals[7];
  float m = -3.4e38f;
#pragma unroll
  for (int i = 0; i < 7; ++i) {
    int t = tid + i * 256;
    float v = (t < T_) ? 2.0f * e[(size_t)b * T_ + t] : -3.4e38f;
    vals[i] = v;
    m = fmaxf(m, v);
  }
  for (int mask = 16; mask; mask >>= 1) m = fmaxf(m, __shfl_xor(m, mask, 32));
  if ((tid & 31) == 0) red[tid >> 5] = m;
  __syncthreads();
  if (tid == 0) {
    float mm = red[0];
    for (int i = 1; i < 8; ++i) mm = fmaxf(mm, red[i]);
    bcast = mm;
  }
  __syncthreads();
  m = bcast;
  float s = 0.f;
#pragma unroll
  for (int i = 0; i < 7; ++i) {
    int t = tid + i * 256;
    if (t < T_) { float ev = __expf(vals[i] - m); vals[i] = ev; s += ev; }
  }
  for (int mask = 16; mask; mask >>= 1) s += __shfl_xor(s, mask, 32);
  __syncthreads();
  if ((tid & 31) == 0) red[tid >> 5] = s;
  __syncthreads();
  if (tid == 0) {
    float ss = 0.f;
    for (int i = 0; i < 8; ++i) ss += red[i];
    bcast = 1.0f / ss;
  }
  __syncthreads();
  float inv = bcast;
#pragma unroll
  for (int i = 0; i < 7; ++i) {
    int t = tid + i * 256;
    if (t < T_) w[(size_t)b * T_ + t] = vals[i] * inv;
  }
}

// ---------------- context: deterministic two-stage reduction ----------------
__global__ __launch_bounds__(256)
void ctx_partial_kernel(const float* __restrict__ enc, const float* __restrict__ w,
                        float* __restrict__ partial) {
  const int chunk = blockIdx.x;                 // 0..7 (200 t each)
  const int d = blockIdx.y * 256 + threadIdx.x;
  const int b = blockIdx.z;
  const int tstart = chunk * 200;
  const float* encb = enc + ((size_t)b * T_ + tstart) * EPROJS + d;
  const float* wb = w + (size_t)b * T_ + tstart;
  float s = 0.f;
  for (int t = 0; t < 200; ++t)
    s += wb[t] * encb[(size_t)t * EPROJS];
  partial[(((size_t)b * 8) + chunk) * EPROJS + d] = s;
}

__global__ __launch_bounds__(256)
void ctx_reduce_kernel(const float* __restrict__ partial, float* __restrict__ c) {
  const int d = blockIdx.x * 256 + threadIdx.x;
  const int b = blockIdx.y;
  float s = 0.f;
#pragma unroll
  for (int k = 0; k < 8; ++k)
    s += partial[(((size_t)b * 8) + k) * EPROJS + d];
  c[(size_t)b * EPROJS + d] = s;
}

// ---------------- launcher ----------------
extern "C" void kernel_launch(void* const* d_in, const int* in_sizes, int n_in,
                              void* d_out, int out_size, void* d_ws, size_t ws_size,
                              hipStream_t stream) {
  (void)in_sizes; (void)n_in; (void)out_size; (void)ws_size;
  const float* enc      = (const float*)d_in[0];
  /* d_in[1] = enc_hs_len : unused by reference math */
  const float* dec_z    = (const float*)d_in[2];
  const float* att_prev = (const float*)d_in[3];
  const float* W_enc    = (const float*)d_in[4];
  const float* b_enc    = (const float*)d_in[5];
  const float* W_dec    = (const float*)d_in[6];
  const float* W_att    = (const float*)d_in[7];
  const float* conv_w   = (const float*)d_in[8];
  const float* gvec_w   = (const float*)d_in[9];
  /* d_in[10] = gvec_b : shift-invariant under softmax -> dropped */

  char* ws = (char*)d_ws;
  float*          convbuf = (float*)(ws);                    // [32][1600][16] f32  = 3,276,800 B
  float*          dvec    = (float*)(ws + 3276800);          // [32][512]     f32  =    65,536 B
  float*          evec    = (float*)(ws + 3342336);          // [32][1600]    f32  =   204,800 B
  unsigned short* wext    = (unsigned short*)(ws + 3547136); // [512][544]    bf16 =   557,056 B
  float*          partial = (float*)(ws + 4104192);          // [32][8][512]  f32  =   524,288 B

  float* c_out = (float*)d_out;                // [32][512]
  float* w_out = (float*)d_out + B_ * ATT;     // [32][1600]

  prep_wext_kernel<<<dim3(512), 256, 0, stream>>>(W_enc, W_att, wext);
  prep_dvec_kernel<<<dim3(512, 32), 256, 0, stream>>>(dec_z, W_dec, b_enc, dvec);
  conv_kernel<<<dim3(7, 32), 256, 0, stream>>>(att_prev, conv_w, convbuf);
  energy_kernel<<<dim3(25, 32), 512, 0, stream>>>(enc, convbuf, dvec, wext, gvec_w, evec);
  softmax_kernel<<<dim3(32), 256, 0, stream>>>(evec, w_out);
  ctx_partial_kernel<<<dim3(8, 2, 32), 256, 0, stream>>>(enc, w_out, partial);
  ctx_reduce_kernel<<<dim3(2, 32), 256, 0, stream>>>(partial, c_out);
}
